// Head_3762391351945
// MI455X (gfx1250) — matrix operand order repaired and checked by reading, not compiled, not run on hardware
//
#include <hip/hip_runtime.h>
#include <math.h>

// Problem constants (match reference): B=512, T=256, C=384, HS=64
#define BATCH 512
#define TT    256
#define CC    384
#define HS    64

typedef __bf16 bf16_t;
typedef bf16_t bf16x16 __attribute__((ext_vector_type(16)));
typedef float  f32x8   __attribute__((ext_vector_type(8)));

union TileAB {
    bf16x16 v;
    bf16_t  e[16];
    uint4   q[2];
};

__device__ __forceinline__ f32x8 zero8() {
    f32x8 z;
#pragma unroll
    for (int i = 0; i < 8; ++i) z[i] = 0.0f;
    return z;
}
__device__ __forceinline__ f32x8 wmma_bf16(const TileAB& a, const TileAB& b, f32x8 c) {
    // D = A(16x32 bf16) * B(32x16 bf16) + C(16x16 f32)
    return __builtin_amdgcn_wmma_f32_16x16x32_bf16(
        false, a.v, false, b.v, (short)0, c, false, false);
}

// LDS layout (bytes)
#define OFF_Q   0        // 256x64 bf16  (32768)
#define OFF_K   32768    // 256x64 bf16  (32768)
#define OFF_VT  65536    // 64x256 bf16, transposed (32768)
#define OFF_WT  98304    // 3 x 64 x 384 bf16, W transposed [mat][n][k] (147456)
#define OFF_M   245760   // 256 f32 column max (1024)
#define OFF_Z   246784   // 256 f32 column 1/Z (1024)
#define OFF_P   247808   // 8 waves x 16x32 bf16 P scratch (8192)
#define SMEM_BYTES 256000

#define SCALE 0.05103103630798288f   // C^-0.5 = 1/sqrt(384)

__global__ __launch_bounds__(256) void head_attn_kernel(
    const float* __restrict__ x,
    const float* __restrict__ Wq,
    const float* __restrict__ Wk,
    const float* __restrict__ Wv,
    float* __restrict__ out)
{
    extern __shared__ char smem[];
    bf16_t* sQ  = (bf16_t*)(smem + OFF_Q);
    bf16_t* sK  = (bf16_t*)(smem + OFF_K);
    bf16_t* sVt = (bf16_t*)(smem + OFF_VT);
    bf16_t* sWt = (bf16_t*)(smem + OFF_WT);
    float*  sM  = (float*)(smem + OFF_M);
    float*  sZi = (float*)(smem + OFF_Z);
    bf16_t* sP  = (bf16_t*)(smem + OFF_P);

    const int tid    = threadIdx.x;
    const int w      = tid >> 5;      // wave id 0..7
    const int lane   = tid & 31;
    const int laneM  = lane & 15;     // column (B/C/D) or row (A) within tile
    const int laneHi = lane >> 4;     // half-wave select
    const int b      = blockIdx.x;    // batch element

    // ---------------- Phase 0: stage W (transposed, bf16) into LDS ----------
    {
        const float* Wm[3] = {Wq, Wk, Wv};
        const int n  = tid & 63;
        const int k0 = (tid >> 6) * 96;   // 4 groups x 96 = 384
#pragma unroll
        for (int mat = 0; mat < 3; ++mat) {
            bf16_t* dst = sWt + mat * (HS * CC) + n * CC;
            const float* src = Wm[mat];
            for (int kk = 0; kk < 96; ++kk) {
                int k = k0 + kk;
                dst[k] = (bf16_t)src[k * HS + n];
            }
        }
    }
    __syncthreads();

    // ---------------- Phase 1: q,k,v = x[b] @ W  (WMMA, bf16) ----------------
    {
        const int m0 = w * 32;
        for (int mt = 0; mt < 2; ++mt) {
            f32x8 acc[3][4];
#pragma unroll
            for (int m = 0; m < 3; ++m)
#pragma unroll
                for (int n = 0; n < 4; ++n) acc[m][n] = zero8();

            const int row = m0 + mt * 16 + laneM;
            const float* xrow = x + ((size_t)b * TT + row) * CC;
            const int lo = laneHi * 8;

            for (int kk = 0; kk < 12; ++kk) {           // K = 384 in 32-chunks
                // A-operand: 16x32 bf16 tile of x (rows of this wave's strip)
                TileAB a;
                const float4 f0 = *(const float4*)(xrow + kk * 32 + lo);
                const float4 f1 = *(const float4*)(xrow + kk * 32 + lo + 4);
                const float4 f2 = *(const float4*)(xrow + kk * 32 + lo + 16);
                const float4 f3 = *(const float4*)(xrow + kk * 32 + lo + 20);
                a.e[0]=(bf16_t)f0.x; a.e[1]=(bf16_t)f0.y; a.e[2]=(bf16_t)f0.z; a.e[3]=(bf16_t)f0.w;
                a.e[4]=(bf16_t)f1.x; a.e[5]=(bf16_t)f1.y; a.e[6]=(bf16_t)f1.z; a.e[7]=(bf16_t)f1.w;
                a.e[8]=(bf16_t)f2.x; a.e[9]=(bf16_t)f2.y; a.e[10]=(bf16_t)f2.z; a.e[11]=(bf16_t)f2.w;
                a.e[12]=(bf16_t)f3.x; a.e[13]=(bf16_t)f3.y; a.e[14]=(bf16_t)f3.z; a.e[15]=(bf16_t)f3.w;
#pragma unroll
                for (int mat = 0; mat < 3; ++mat) {
#pragma unroll
                    for (int nt = 0; nt < 4; ++nt) {
                        TileAB bt;
                        const bf16_t* wp = sWt + mat * (HS * CC)
                            + (nt * 16 + laneM) * CC + kk * 32 + laneHi * 8;
                        bt.q[0] = *(const uint4*)(wp);
                        bt.q[1] = *(const uint4*)(wp + 16);
                        acc[mat][nt] = wmma_bf16(a, bt, acc[mat][nt]);
                    }
                }
            }
            // write results to LDS (v stored transposed)
#pragma unroll
            for (int mat = 0; mat < 3; ++mat)
#pragma unroll
                for (int nt = 0; nt < 4; ++nt)
#pragma unroll
                    for (int r = 0; r < 8; ++r) {
                        const int rr = m0 + mt * 16 + r + 8 * laneHi;
                        const int cc = nt * 16 + laneM;
                        const bf16_t val = (bf16_t)acc[mat][nt][r];
                        if (mat == 0)      sQ[rr * HS + cc] = val;
                        else if (mat == 1) sK[rr * HS + cc] = val;
                        else               sVt[cc * TT + rr] = val;
                    }
        }
    }
    __syncthreads();

    // ---------------- Phase 2: per-KEY-column online max/sum of causal S ----
    for (int i = 0; i < 2; ++i) {
        const int kt   = 2 * w + i;          // key tile owned by this wave
        const int colg = kt * 16 + laneM;    // global key column
        float m = -INFINITY, Z = 0.0f;
        for (int qt = kt; qt < 16; ++qt) {   // only causal (lower) tiles
            f32x8 s = zero8();
#pragma unroll
            for (int h0 = 0; h0 < HS; h0 += 32) {
                TileAB a;                    // q rows of tile qt
                const bf16_t* qp = sQ + (qt * 16 + laneM) * HS + h0 + laneHi * 8;
                a.q[0] = *(const uint4*)qp;
                a.q[1] = *(const uint4*)(qp + 16);
                TileAB bt;                   // k^T: lane col = key row
                const bf16_t* kp = sK + (kt * 16 + laneM) * HS + h0 + laneHi * 8;
                bt.q[0] = *(const uint4*)kp;
                bt.q[1] = *(const uint4*)(kp + 16);
                s = wmma_bf16(a, bt, s);
            }
            const int rowbase = qt * 16 + 8 * laneHi;
            float sv[8], tmax = -INFINITY;
#pragma unroll
            for (int r = 0; r < 8; ++r) {
                const bool ok = (rowbase + r) >= colg;
                sv[r] = ok ? s[r] * SCALE : -INFINITY;
                tmax = fmaxf(tmax, sv[r]);
            }
            tmax = fmaxf(tmax, __shfl_xor(tmax, 16, 32)); // combine row halves
            const float mnew = fmaxf(m, tmax);
            const float fac  = __expf(m - mnew);          // 0 when m==-inf
            float ps = 0.0f;
#pragma unroll
            for (int r = 0; r < 8; ++r) ps += __expf(sv[r] - mnew);
            ps += __shfl_xor(ps, 16, 32);
            Z = Z * fac + ps;
            m = mnew;
        }
        if (lane < 16) { sM[colg] = m; sZi[colg] = 1.0f / Z; }
    }
    __syncthreads();

    // ---------------- Phase 2.5: fold 1/Z into V' rows ----------------------
    for (int i2 = tid; i2 < HS * TT; i2 += 256) {
        const int key = i2 & 255;
        sVt[i2] = (bf16_t)((float)sVt[i2] * sZi[key]);
    }
    __syncthreads();

    // ---------------- Phase 3: out = exp(S - m_col) @ V' --------------------
    bf16_t* sPw = sP + w * 512;              // 16x32 bf16 per-wave scratch
    for (int i = 0; i < 2; ++i) {
        const int qt = 2 * w + i;
        f32x8 o[4];
#pragma unroll
        for (int nt = 0; nt < 4; ++nt) o[nt] = zero8();
        const int rowbase = qt * 16 + 8 * laneHi;

        for (int p = 0; p <= (qt >> 1); ++p) {   // key-tile pairs (K=32 step)
#pragma unroll
            for (int jj = 0; jj < 2; ++jj) {
                const int kt = 2 * p + jj;
                if (kt <= qt) {
                    f32x8 s = zero8();
#pragma unroll
                    for (int h0 = 0; h0 < HS; h0 += 32) {
                        TileAB a;
                        const bf16_t* qp = sQ + (qt * 16 + laneM) * HS + h0 + laneHi * 8;
                        a.q[0] = *(const uint4*)qp;
                        a.q[1] = *(const uint4*)(qp + 16);
                        TileAB bt;
                        const bf16_t* kp = sK + (kt * 16 + laneM) * HS + h0 + laneHi * 8;
                        bt.q[0] = *(const uint4*)kp;
                        bt.q[1] = *(const uint4*)(kp + 16);
                        s = wmma_bf16(a, bt, s);
                    }
                    const int   colg2 = kt * 16 + laneM;
                    const float mcol  = sM[colg2];
#pragma unroll
                    for (int r = 0; r < 8; ++r) {
                        const bool ok = (rowbase + r) >= colg2;
                        const float pv = ok ? __expf(s[r] * SCALE - mcol) : 0.0f;
                        sPw[(r + 8 * laneHi) * 32 + jj * 16 + laneM] = (bf16_t)pv;
                    }
                } else {
#pragma unroll
                    for (int r = 0; r < 8; ++r)
                        sPw[(r + 8 * laneHi) * 32 + jj * 16 + laneM] = (bf16_t)0.0f;
                }
            }
            // A-operand = P tile (16x32) from per-wave LDS (DS ops are
            // in-order within a wave, so store->load needs no barrier)
            TileAB a;
            const bf16_t* pp = sPw + laneM * 32 + laneHi * 8;
            a.q[0] = *(const uint4*)pp;
            a.q[1] = *(const uint4*)(pp + 16);
#pragma unroll
            for (int nt = 0; nt < 4; ++nt) {
                TileAB bt;                   // V' (transposed in LDS): contiguous K
                const bf16_t* vp = sVt + (nt * 16 + laneM) * TT + p * 32 + laneHi * 8;
                bt.q[0] = *(const uint4*)vp;
                bt.q[1] = *(const uint4*)(vp + 16);
                o[nt] = wmma_bf16(a, bt, o[nt]);
            }
        }
        // store output tile (f32)
#pragma unroll
        for (int nt = 0; nt < 4; ++nt)
#pragma unroll
            for (int r = 0; r < 8; ++r) {
                const int rr = rowbase + r;
                out[((size_t)b * TT + rr) * HS + nt * 16 + laneM] = o[nt][r];
            }
    }
}

extern "C" void kernel_launch(void* const* d_in, const int* in_sizes, int n_in,
                              void* d_out, int out_size, void* d_ws, size_t ws_size,
                              hipStream_t stream) {
    const float* x  = (const float*)d_in[0];
    const float* Wq = (const float*)d_in[1];
    const float* Wk = (const float*)d_in[2];
    const float* Wv = (const float*)d_in[3];
    float* out = (float*)d_out;
    (void)in_sizes; (void)n_in; (void)out_size; (void)d_ws; (void)ws_size;

    head_attn_kernel<<<dim3(BATCH), dim3(256), SMEM_BYTES, stream>>>(x, Wq, Wk, Wv, out);
}
